// LoRALinearOn4bit_32100585571129
// MI455X (gfx1250) — matrix operand order repaired and checked by reading, not compiled, not run on hardware
//
#include <hip/hip_runtime.h>
#include <hip/hip_bf16.h>

// ---------------------------------------------------------------------------
// LoRA-on-linear folded into a single bf16 WMMA GEMM for gfx1250 (MI455X).
//   out = x @ (W + s*B@A)^T + b
// - Prep kernel folds LoRA into W_eff (bf16, in d_ws).
// - GEMM: 128x256 block tile, BK=32, 8 waves @ 64x64 each (16 WMMAs/wave/step).
// - W tile moved by the Tensor Data Mover (tensor_load_to_lds, double-buffered,
//   s_wait_tensorcnt). x tile staged via registers with f32->bf16 conversion.
// ---------------------------------------------------------------------------

#define DIN   4096
#define DOUT  4096
#define MTOT  8192
#define LORAR 16
#define SCALE 2.0f

typedef __bf16 bf16t;
typedef __attribute__((ext_vector_type(16))) __bf16 v16bf;
typedef __attribute__((ext_vector_type(8)))  __bf16 v8bf;
typedef __attribute__((ext_vector_type(4)))  __bf16 v4bf;
typedef __attribute__((ext_vector_type(8)))  float  v8f;
typedef __attribute__((ext_vector_type(4)))  unsigned int v4u;
typedef __attribute__((ext_vector_type(8)))  int v8i;
typedef __attribute__((ext_vector_type(4)))  int v4i;

#define BM 128
#define BN 256
#define BK 32
#define BKP 40                 // padded LDS row stride (elements): 80B = 64B data + 16B pad
#define KT (DIN / BK)          // 128 K-tiles

// --------------------------- prep: W_eff = bf16(W + s * B@A) ----------------
__global__ __launch_bounds__(256) void fold_weights_kernel(
    const float* __restrict__ W, const float* __restrict__ A,
    const float* __restrict__ Bm, bf16t* __restrict__ Weff)
{
  int gid = blockIdx.x * 256 + threadIdx.x;      // one thread per 4 elements
  int o = gid >> 10;                             // gid / (DIN/4)
  int d = (gid & 1023) << 2;

  float ax = 0.f, ay = 0.f, az = 0.f, aw = 0.f;
#pragma unroll
  for (int r = 0; r < LORAR; ++r) {
    float br = Bm[o * LORAR + r];
    float4 av = *(const float4*)(A + (size_t)r * DIN + d);
    ax += br * av.x; ay += br * av.y; az += br * av.z; aw += br * av.w;
  }
  float4 w = *(const float4*)(W + (size_t)o * DIN + d);
  v4bf e;
  e[0] = (__bf16)(w.x + SCALE * ax);
  e[1] = (__bf16)(w.y + SCALE * ay);
  e[2] = (__bf16)(w.z + SCALE * az);
  e[3] = (__bf16)(w.w + SCALE * aw);
  *(v4bf*)(Weff + (size_t)o * DIN + d) = e;
}

// --------------------------- helpers ---------------------------------------
static __device__ __forceinline__ v8bf cvt8(float4 a, float4 b) {
  v8bf r;
  r[0] = (__bf16)a.x; r[1] = (__bf16)a.y; r[2] = (__bf16)a.z; r[3] = (__bf16)a.w;
  r[4] = (__bf16)b.x; r[5] = (__bf16)b.y; r[6] = (__bf16)b.z; r[7] = (__bf16)b.w;
  return r;
}
static __device__ __forceinline__ v16bf combine16(v8bf lo8, v8bf hi8) {
  union { v16bf v; v8bf h[2]; } u;
  u.h[0] = lo8; u.h[1] = hi8;
  return u.v;
}

// TDM: load a BN x BK bf16 tile (row stride DIN elements) into LDS with
// hardware padding: 16B pad after every 64B row -> 80B LDS row stride (= BKP).
// D# bitfields per CDNA5 ISA 8.3/8.4. 6-arg builtin form (clang-23 toolchain).
static __device__ __forceinline__ void tdm_load_w_tile(unsigned lds_off,
                                                       const bf16t* gsrc) {
  unsigned long long ga = (unsigned long long)(size_t)gsrc;
  v4u g0;
  g0[0] = 1u;                                        // count=1, load, user D#
  g0[1] = lds_off;                                   // lds_addr (bytes)
  g0[2] = (unsigned)(ga & 0xFFFFFFFFu);              // global_addr[31:0]
  g0[3] = (unsigned)((ga >> 32) & 0x01FFFFFFu)       // global_addr[56:32]
        | 0x80000000u;                               // type=2 ("image")
  v8i g1;
  g1[0] = (int)((1u << 16)        // data_size = 2 bytes
              | (1u << 20)        // pad_enable
              | (3u << 22)        // pad_interval: every 16 DWORDs (64B)
              | (3u << 25));      // pad_amount: 4 DWORDs (16B)
  g1[1] = (int)((unsigned)(DIN & 0xFFFF) << 16);     // tensor_dim0[15:0] @ [63:48]
  g1[2] = (int)(((unsigned)DIN >> 16)                // tensor_dim0[31:16]
              | ((unsigned)(DOUT & 0xFFFF) << 16));  // tensor_dim1[15:0]
  g1[3] = (int)(((unsigned)DOUT >> 16)               // tensor_dim1[31:16]
              | ((unsigned)BK << 16));               // tile_dim0 = 32
  g1[4] = (int)(unsigned)BN;                         // tile_dim1 = 256, tile_dim2 = 0
  g1[5] = (int)(unsigned)DIN;                        // tensor_dim0_stride[31:0]
  g1[6] = 0;                                         // stride hi, dim1_stride lo
  g1[7] = 0;
  v4i gz4  = {0, 0, 0, 0};
  v8i gz8  = {0, 0, 0, 0, 0, 0, 0, 0};
  __builtin_amdgcn_tensor_load_to_lds(g0, g1, gz4, gz4, gz8, 0);
}

// --------------------------- main GEMM --------------------------------------
__global__ __launch_bounds__(256) void lora_gemm_kernel(
    const float* __restrict__ X, const bf16t* __restrict__ Weff,
    const float* __restrict__ bias, float* __restrict__ Out)
{
  __shared__ bf16t lA[2][BM * BKP];   // x tile,  128x32 bf16, double-buffered
  __shared__ bf16t lB[2][BN * BKP];   // W tile,  256x32 bf16, double-buffered (TDM)

  const int tid   = threadIdx.x;
  const int lane  = tid & 31;
  const int wave  = tid >> 5;       // 0..7
  const int waveM = wave >> 2;      // 0..1 -> 64-row strip
  const int waveN = wave & 3;       // 0..3 -> 64-col strip
  const int tileM = blockIdx.y * BM;
  const int tileN = blockIdx.x * BN;

  const int l15 = lane & 15;
  const int hic = lane >> 4;        // 0 for lanes 0-15, 1 for lanes 16-31

  // x staging assignment: 2 threads per row, 16 f32 each
  const int gRow = tid >> 1;                 // 0..127
  const int gCol = (tid & 1) * 16;           // 0 or 16
  const float* xg = X + (size_t)(tileM + gRow) * DIN + gCol;
  const bf16t* wTile = Weff + (size_t)tileN * DIN;   // block-uniform, for TDM

  v8f acc[4][4] = {};                        // 16 C fragments (64x64 per wave)

  float4 xr0, xr1, xr2, xr3;                 // staged x (f32)

#define LOAD_X(kt)                                               \
  {                                                              \
    const float* xp = xg + (kt) * BK;                            \
    xr0 = *(const float4*)(xp);                                  \
    xr1 = *(const float4*)(xp + 4);                              \
    xr2 = *(const float4*)(xp + 8);                              \
    xr3 = *(const float4*)(xp + 12);                             \
  }

#define STORE_X(buf)                                             \
  {                                                              \
    *(v8bf*)&lA[buf][gRow * BKP + gCol]     = cvt8(xr0, xr1);    \
    *(v8bf*)&lA[buf][gRow * BKP + gCol + 8] = cvt8(xr2, xr3);    \
  }

  // ---- prologue: tile 0 ----
  if (wave == 0) {
    tdm_load_w_tile((unsigned)(size_t)&lB[0][0], wTile);
    __builtin_amdgcn_s_wait_tensorcnt(0);
  }
  LOAD_X(0);
  STORE_X(0);
  __syncthreads();

  for (int kt = 0; kt < KT; ++kt) {
    const int cur = kt & 1;
    const int nxt = cur ^ 1;

    if (kt + 1 < KT) {
      if (wave == 0)
        tdm_load_w_tile((unsigned)(size_t)&lB[nxt][0], wTile + (kt + 1) * BK);
      LOAD_X(kt + 1);
      STORE_X(nxt);
      if (kt + 2 < KT)
        __builtin_prefetch(xg + (kt + 2) * BK, 0, 0);
    }

    // ---- fragments from LDS (layouts per CDNA5 ISA 7.12.2) ----
    // B (32x16, KxN): lane = column n; lanes 0-15 K {0..7,16..23}; lanes 16-31 K {8..15,24..31}
    v16bf bfr[4];
#pragma unroll
    for (int g = 0; g < 4; ++g) {
      const bf16t* base = &lB[cur][(waveN * 64 + g * 16 + l15) * BKP] + hic * 8;
      bfr[g] = combine16(*(const v8bf*)(base), *(const v8bf*)(base + 16));
    }
    // A (16x32, MxK): lanes 0-15 K {0..7,16..23}; lanes 16-31 K {8..15,24..31}
#pragma unroll
    for (int f = 0; f < 4; ++f) {
      const bf16t* abase = &lA[cur][(waveM * 64 + f * 16 + l15) * BKP];
      v16bf af = combine16(*(const v8bf*)(abase + hic * 8),
                           *(const v8bf*)(abase + 16 + hic * 8));
#pragma unroll
      for (int g = 0; g < 4; ++g)
        acc[f][g] = __builtin_amdgcn_wmma_f32_16x16x32_bf16(
            false, af, false, bfr[g], (short)0, acc[f][g], false, false);
    }

    if (wave == 0)
      __builtin_amdgcn_s_wait_tensorcnt(0);   // TDM tile (kt+1) landed
    __syncthreads();
  }

  // ---- epilogue: C/D layout -> global, fused bias ----
  // VGPR i: lanes 0-15 -> M=i, N=lane; lanes 16-31 -> M=8+i, N=lane-16
#pragma unroll
  for (int g = 0; g < 4; ++g) {
    const int n  = tileN + waveN * 64 + g * 16 + l15;
    const float bv = bias[n];
#pragma unroll
    for (int f = 0; f < 4; ++f) {
      const int m0 = tileM + waveM * 64 + f * 16 + hic * 8;
      v8f c = acc[f][g];
#pragma unroll
      for (int i = 0; i < 8; ++i)
        Out[(size_t)(m0 + i) * DOUT + n] = c[i] + bv;
    }
  }
#undef LOAD_X
#undef STORE_X
}

// --------------------------- launcher ---------------------------------------
extern "C" void kernel_launch(void* const* d_in, const int* in_sizes, int n_in,
                              void* d_out, int out_size, void* d_ws, size_t ws_size,
                              hipStream_t stream) {
  (void)in_sizes; (void)n_in; (void)out_size; (void)ws_size;

  const float* x    = (const float*)d_in[0];  // [4,2048,4096]
  const float* W    = (const float*)d_in[1];  // [4096,4096]
  const float* b    = (const float*)d_in[2];  // [4096]
  const float* A    = (const float*)d_in[3];  // [16,4096]
  const float* Bm   = (const float*)d_in[4];  // [4096,16]
  float*       out  = (float*)d_out;          // [4,2048,4096]
  bf16t*       Weff = (bf16t*)d_ws;           // 32 MiB bf16 scratch

  // 1) fold LoRA into the weight, convert to bf16
  dim3 gridFold((DOUT * (DIN / 4)) / 256);
  fold_weights_kernel<<<gridFold, 256, 0, stream>>>(W, A, Bm, Weff);

  // 2) single bf16 WMMA GEMM with fused bias, TDM-fed weight tiles
  dim3 gridGemm(DOUT / BN, MTOT / BM);
  lora_gemm_kernel<<<gridGemm, 256, 0, stream>>>(x, Weff, b, out);
}
